// LayerMultiheadSelfAttention_73280732004944
// MI455X (gfx1250) — matrix operand order repaired and checked by reading, compiled, not run on hardware
//
#include <hip/hip_runtime.h>
#include <hip/hip_bf16.h>

// MHSA block for B=2, S=2048, D=1024, H=16, HD=64 on gfx1250 (CDNA5, wave32).
// All matmuls via v_wmma_f32_16x16x32_bf16 (f32 accumulate), flash-softmax in
// registers using the C/D VGPR layout, P transposed C->A layout through LDS.
// GEMMs: 8 waves / 128x64 tile, double-buffered LDS weight tiles (1 barrier
// per K-step). Attention: wave-private LDS P tile with wave-local DS fence.

typedef __attribute__((ext_vector_type(16))) __bf16 v16bf;
typedef __attribute__((ext_vector_type(8)))  __bf16 v8bf;
typedef __attribute__((ext_vector_type(8)))  float  v8f;

#define BB  2
#define SS  2048
#define DD  1024
#define HH  16
#define HDD 64
#define MM  (BB * SS)   // 4096 rows

__device__ __forceinline__ v16bf pack16(v8bf a, v8bf b) {
  v16bf r;
#pragma unroll
  for (int i = 0; i < 8; ++i) { r[i] = a[i]; r[i + 8] = b[i]; }
  return r;
}

__device__ __forceinline__ v8f wmma_bf16(v16bf a, v16bf b, v8f c) {
  // D = A(16x32 bf16) x B(32x16 bf16) + C(16x16 f32)
  return __builtin_amdgcn_wmma_f32_16x16x32_bf16(false, a, false, b,
                                                 (short)0, c, false, false);
}

// Wave-local LDS fence: orders per-wave cross-lane ds_store -> ds_load without
// a workgroup barrier (LDS ops are in-order per wave; this pins the compiler
// schedule and drains DScnt).
__device__ __forceinline__ void wave_lds_fence() {
  __builtin_amdgcn_wave_barrier();
  asm volatile("s_wait_dscnt 0x0" ::: "memory");
  __builtin_amdgcn_wave_barrier();
}

// ---------------------------------------------------------------- converts --
__global__ __launch_bounds__(256) void cvt_f32_bf16(const float* __restrict__ src,
                                                    __bf16* __restrict__ dst,
                                                    int n4) {
  int i = blockIdx.x * 256 + threadIdx.x;
  if (i >= n4) return;
  float4 f = ((const float4*)src)[i];
  __bf16 o[4] = { (__bf16)f.x, (__bf16)f.y, (__bf16)f.z, (__bf16)f.w };
  *(uint2*)(dst + (size_t)i * 4) = *(uint2*)o;  // 8B packed store
}

// W[k][n] f32 -> Wt[n][k] bf16  (so B-fragments read contiguous K)
__global__ __launch_bounds__(256) void cvt_w_T(const float* __restrict__ w,
                                               __bf16* __restrict__ wt) {
  int i = blockIdx.x * 256 + threadIdx.x;     // 0 .. 1048575
  int k = i >> 10, n = i & 1023;
  wt[(size_t)n * DD + k] = (__bf16)w[i];
}

// -------------------------------------------------------------- QKV GEMM ----
// grid (N/64=16, M/128=32, 3), block 256 (8 waves; 16 rows x 64 cols per wave)
__global__ __launch_bounds__(256) void gemm_qkv(const __bf16* __restrict__ X,
                                                const __bf16* __restrict__ Wbase,
                                                __bf16* __restrict__ Qo,
                                                __bf16* __restrict__ Ko,
                                                __bf16* __restrict__ Vo) {
  __shared__ __bf16 ldsw[2][64 * 40];         // double-buffered 64x32 W^T tile
  const int which = blockIdx.z;
  const __bf16* WT = Wbase + (size_t)which * DD * DD;
  const int n0   = blockIdx.x * 64;
  const int wave = threadIdx.x >> 5;
  const int lane = threadIdx.x & 31;
  const int lh = lane & 15, hs = lane >> 4;
  const int m0 = blockIdx.y * 128 + wave * 16;
  const int srow = threadIdx.x >> 2;          // staging: 64 rows x 4 quarters
  const int sq   = threadIdx.x & 3;

  v8f acc[4];
#pragma unroll
  for (int j = 0; j < 4; ++j) { v8f z = {0,0,0,0,0,0,0,0}; acc[j] = z; }

  const __bf16* gW = WT + (size_t)(n0 + srow) * DD + sq * 8;  // + k0 later
  const __bf16* gA = X + (size_t)(m0 + lh) * DD + hs * 8;     // + k0 later

  // preload first tile
  *(v8bf*)(&ldsw[0][srow * 40 + sq * 8]) = *(const v8bf*)(gW);
  __syncthreads();

  for (int k0 = 0; k0 < DD; k0 += 32) {
    const int buf = (k0 >> 5) & 1;
    const bool has_next = (k0 + 32) < DD;
    v8bf wnext;
    if (has_next) wnext = *(const v8bf*)(gW + k0 + 32);   // overlap with WMMAs

    // A fragment: row = m0+lh, K chunks {0..7,16..23} (+8 for upper half-wave)
    const __bf16* ar = gA + k0;
    v16bf A = pack16(*(const v8bf*)ar, *(const v8bf*)(ar + 16));

    // gather all 4 B fragments first -> one DS wait, 4 back-to-back WMMAs
    v16bf Bf[4];
#pragma unroll
    for (int j = 0; j < 4; ++j) {
      const __bf16* br = &ldsw[buf][(j * 16 + lh) * 40 + hs * 8];
      Bf[j] = pack16(*(const v8bf*)br, *(const v8bf*)(br + 16));
    }
#pragma unroll
    for (int j = 0; j < 4; ++j) acc[j] = wmma_bf16(A, Bf[j], acc[j]);

    if (has_next)
      *(v8bf*)(&ldsw[buf ^ 1][srow * 40 + sq * 8]) = wnext;
    __syncthreads();   // publish next tile / retire reads of this one
  }

  // epilogue: scatter into attention-friendly layouts
#pragma unroll
  for (int j = 0; j < 4; ++j) {
#pragma unroll
    for (int r = 0; r < 8; ++r) {
      int gm = m0 + r + hs * 8;               // C/D layout row
      int n  = n0 + j * 16 + lh;              // C/D layout col
      int b  = gm >> 11, s = gm & (SS - 1);
      int h  = n >> 6,  hd = n & 63;
      float v = acc[j][r];
      size_t bh = (size_t)(b * HH + h);
      if (which == 0)       Qo[(bh * SS + s) * HDD + hd] = (__bf16)(v * 0.25f); // 1/sqrt(H)
      else if (which == 1)  Ko[(bh * SS + s) * HDD + hd] = (__bf16)v;
      else                  Vo[(bh * HDD + hd) * SS + s] = (__bf16)v;           // V^T
    }
  }
}

// ------------------------------------------------------------- attention ----
// grid (S/64=32, B*H=32), block 128. Each wave: 16 q-rows, full HD=64,
// flash-style online softmax over 32-key steps. No workgroup barriers.
__global__ __launch_bounds__(128) void attn_kernel(const __bf16* __restrict__ Qb,
                                                   const __bf16* __restrict__ Kb,
                                                   const __bf16* __restrict__ Vtb,
                                                   __bf16* __restrict__ Ctx) {
  __shared__ __bf16 ldsp[4 * 16 * 40];        // per-wave 16x32 P tile (pad 40)
  const int wave = threadIdx.x >> 5;
  const int lane = threadIdx.x & 31;
  const int lh = lane & 15, hs = lane >> 4;
  const int bh = blockIdx.y;
  const int q0 = blockIdx.x * 64 + wave * 16;
  const __bf16* Qp = Qb  + (size_t)bh * SS * HDD;
  const __bf16* Kp = Kb  + (size_t)bh * SS * HDD;
  const __bf16* Vp = Vtb + (size_t)bh * HDD * SS;
  __bf16* pbuf = &ldsp[wave * 16 * 40];

  // Q A-fragments (hd 0..31 and 32..63), reused for all keys
  v16bf QA[2];
#pragma unroll
  for (int t = 0; t < 2; ++t) {
    const __bf16* qr = Qp + (size_t)(q0 + lh) * HDD + t * 32 + hs * 8;
    QA[t] = pack16(*(const v8bf*)qr, *(const v8bf*)(qr + 16));
  }

  v8f O[4];
#pragma unroll
  for (int j = 0; j < 4; ++j) { v8f z = {0,0,0,0,0,0,0,0}; O[j] = z; }
  float mrow[8], lrow[8];
#pragma unroll
  for (int r = 0; r < 8; ++r) { mrow[r] = -3.0e38f; lrow[r] = 0.f; }

  for (int kv = 0; kv < SS; kv += 32) {
    // prefetch K tile two steps ahead (32 lanes cover 32 rows of 128B)
    if (kv + 64 < SS)
      __builtin_prefetch(Kp + (size_t)(kv + 64 + lane) * HDD, 0, 1);

    // ---- scores: two 16x16 tiles, contraction over HD=64 (2 WMMAs each) ----
    v16bf KB[4];
#pragma unroll
    for (int kt = 0; kt < 2; ++kt) {
      const __bf16* kr = Kp + (size_t)(kv + kt * 16 + lh) * HDD + hs * 8;
      KB[kt * 2 + 0] = pack16(*(const v8bf*)kr,        *(const v8bf*)(kr + 16)); // hd 0..31
      KB[kt * 2 + 1] = pack16(*(const v8bf*)(kr + 32), *(const v8bf*)(kr + 48)); // hd 32..63
    }
    v8f sc[2];
#pragma unroll
    for (int kt = 0; kt < 2; ++kt) {
      v8f z = {0,0,0,0,0,0,0,0};
      v8f t0 = wmma_bf16(QA[0], KB[kt * 2 + 0], z);
      sc[kt]  = wmma_bf16(QA[1], KB[kt * 2 + 1], t0);
    }

    // ---- row max across 32 keys (rows live across 16-lane groups) ----
    float mt[8];
#pragma unroll
    for (int r = 0; r < 8; ++r) mt[r] = fmaxf(sc[0][r], sc[1][r]);
#pragma unroll
    for (int msk = 1; msk < 16; msk <<= 1)
#pragma unroll
      for (int r = 0; r < 8; ++r) mt[r] = fmaxf(mt[r], __shfl_xor(mt[r], msk, 32));

    float alpha[8];
#pragma unroll
    for (int r = 0; r < 8; ++r) {
      float mn = fmaxf(mrow[r], mt[r]);
      alpha[r] = __expf(mrow[r] - mn);
      mrow[r]  = mn;
    }

    float psum[8];
#pragma unroll
    for (int r = 0; r < 8; ++r) {
      float p0 = __expf(sc[0][r] - mrow[r]);
      float p1 = __expf(sc[1][r] - mrow[r]);
      psum[r] = p0 + p1;
      int row = r + hs * 8;                    // C-layout row
      pbuf[row * 40 + lh]      = (__bf16)p0;   // C-layout -> row-major LDS
      pbuf[row * 40 + 16 + lh] = (__bf16)p1;
    }
    wave_lds_fence();                          // per-wave store->load ordering

#pragma unroll
    for (int msk = 1; msk < 16; msk <<= 1)
#pragma unroll
      for (int r = 0; r < 8; ++r) psum[r] += __shfl_xor(psum[r], msk, 32);
#pragma unroll
    for (int r = 0; r < 8; ++r) lrow[r] = lrow[r] * alpha[r] + psum[r];

    // rescale running output
#pragma unroll
    for (int j = 0; j < 4; ++j)
#pragma unroll
      for (int r = 0; r < 8; ++r) O[j][r] *= alpha[r];

    // P as A-fragment (16x32 over keys)
    const __bf16* pr = pbuf + lh * 40 + hs * 8;
    v16bf PA = pack16(*(const v8bf*)pr, *(const v8bf*)(pr + 16));

    // ---- O += P x V : 4 WMMAs over hd tiles, V^T gives contiguous keys ----
    v16bf VB[4];
#pragma unroll
    for (int j = 0; j < 4; ++j) {
      const __bf16* vr = Vp + (size_t)(j * 16 + lh) * SS + kv + hs * 8;
      VB[j] = pack16(*(const v8bf*)vr, *(const v8bf*)(vr + 16));
    }
#pragma unroll
    for (int j = 0; j < 4; ++j) O[j] = wmma_bf16(PA, VB[j], O[j]);
  }

  // normalize + write ctx back as [m][h*64+hd] bf16 for the out-proj GEMM
  const int b = bh >> 4, h = bh & 15;
#pragma unroll
  for (int j = 0; j < 4; ++j)
#pragma unroll
    for (int r = 0; r < 8; ++r) {
      int qrow = q0 + r + hs * 8;
      int gm   = b * SS + qrow;
      int col  = h * HDD + j * 16 + lh;
      Ctx[(size_t)gm * DD + col] = (__bf16)(O[j][r] / lrow[r]);
    }
}

// --------------------------------------------------- out-proj + bias + res --
// grid (N/64=16, M/128=32), block 256
__global__ __launch_bounds__(256) void gemm_out(const __bf16* __restrict__ A,
                                                const __bf16* __restrict__ WoT,
                                                const float* __restrict__ bo,
                                                const float* __restrict__ resid,
                                                float* __restrict__ outp) {
  __shared__ __bf16 ldsw[2][64 * 40];
  const int n0   = blockIdx.x * 64;
  const int wave = threadIdx.x >> 5;
  const int lane = threadIdx.x & 31;
  const int lh = lane & 15, hs = lane >> 4;
  const int m0 = blockIdx.y * 128 + wave * 16;
  const int srow = threadIdx.x >> 2;
  const int sq   = threadIdx.x & 3;

  v8f acc[4];
#pragma unroll
  for (int j = 0; j < 4; ++j) { v8f z = {0,0,0,0,0,0,0,0}; acc[j] = z; }

  const __bf16* gW = WoT + (size_t)(n0 + srow) * DD + sq * 8;
  const __bf16* gA = A + (size_t)(m0 + lh) * DD + hs * 8;

  *(v8bf*)(&ldsw[0][srow * 40 + sq * 8]) = *(const v8bf*)(gW);
  __syncthreads();

  for (int k0 = 0; k0 < DD; k0 += 32) {
    const int buf = (k0 >> 5) & 1;
    const bool has_next = (k0 + 32) < DD;
    v8bf wnext;
    if (has_next) wnext = *(const v8bf*)(gW + k0 + 32);

    const __bf16* ar = gA + k0;
    v16bf Af = pack16(*(const v8bf*)ar, *(const v8bf*)(ar + 16));

    v16bf Bf[4];
#pragma unroll
    for (int j = 0; j < 4; ++j) {
      const __bf16* br = &ldsw[buf][(j * 16 + lh) * 40 + hs * 8];
      Bf[j] = pack16(*(const v8bf*)br, *(const v8bf*)(br + 16));
    }
#pragma unroll
    for (int j = 0; j < 4; ++j) acc[j] = wmma_bf16(Af, Bf[j], acc[j]);

    if (has_next)
      *(v8bf*)(&ldsw[buf ^ 1][srow * 40 + sq * 8]) = wnext;
    __syncthreads();
  }

#pragma unroll
  for (int j = 0; j < 4; ++j) {
    int n = n0 + j * 16 + lh;
    float bias = bo[n];
#pragma unroll
    for (int r = 0; r < 8; ++r) {
      int gm = m0 + r + hs * 8;
      size_t idx = (size_t)gm * DD + n;
      outp[idx] = acc[j][r] + bias + resid[idx];
    }
  }
}

// ------------------------------------------------------------- layernorm ----
__global__ __launch_bounds__(256) void layernorm_kernel(const float* __restrict__ x,
                                                        const float* __restrict__ gamma,
                                                        const float* __restrict__ beta,
                                                        float* __restrict__ outp) {
  __shared__ float red0[8], red1[8];
  __shared__ float mv[2];
  int row = blockIdx.x;
  float4 v = ((const float4*)(x + (size_t)row * DD))[threadIdx.x];
  float s  = v.x + v.y + v.z + v.w;
  float s2 = v.x * v.x + v.y * v.y + v.z * v.z + v.w * v.w;
#pragma unroll
  for (int msk = 1; msk < 32; msk <<= 1) {
    s  += __shfl_xor(s,  msk, 32);
    s2 += __shfl_xor(s2, msk, 32);
  }
  int w = threadIdx.x >> 5, ln = threadIdx.x & 31;
  if (ln == 0) { red0[w] = s; red1[w] = s2; }
  __syncthreads();
  if (threadIdx.x == 0) {
    float ts = 0.f, ts2 = 0.f;
#pragma unroll
    for (int i = 0; i < 8; ++i) { ts += red0[i]; ts2 += red1[i]; }
    float mu  = ts * (1.0f / DD);
    float var = ts2 * (1.0f / DD) - mu * mu;
    mv[0] = mu;
    mv[1] = rsqrtf(var + 1e-6f);
  }
  __syncthreads();
  float mu = mv[0], rs = mv[1];
  float4 g = ((const float4*)gamma)[threadIdx.x];
  float4 bt = ((const float4*)beta)[threadIdx.x];
  float4 o;
  o.x = (v.x - mu) * rs * g.x + bt.x;
  o.y = (v.y - mu) * rs * g.y + bt.y;
  o.z = (v.z - mu) * rs * g.z + bt.z;
  o.w = (v.w - mu) * rs * g.w + bt.w;
  ((float4*)(outp + (size_t)row * DD))[threadIdx.x] = o;
}

// ----------------------------------------------------------------- launch ---
extern "C" void kernel_launch(void* const* d_in, const int* in_sizes, int n_in,
                              void* d_out, int out_size, void* d_ws, size_t ws_size,
                              hipStream_t stream) {
  (void)in_sizes; (void)n_in; (void)out_size; (void)ws_size;
  const float* x   = (const float*)d_in[0];
  const float* wq  = (const float*)d_in[1];
  const float* wk  = (const float*)d_in[2];
  const float* wv  = (const float*)d_in[3];
  const float* wo  = (const float*)d_in[4];
  const float* bo  = (const float*)d_in[5];
  const float* gam = (const float*)d_in[6];
  const float* bet = (const float*)d_in[7];
  float* outp = (float*)d_out;

  char* ws = (char*)d_ws;                       // 64 MiB layout
  __bf16* Xb  = (__bf16*)(ws + (0ull  << 20));  // [4096][1024]        8 MiB
  __bf16* Wt  = (__bf16*)(ws + (8ull  << 20));  // 4x W^T [1024][1024] 8 MiB
  __bf16* Qb  = (__bf16*)(ws + (16ull << 20));  // [B,H,S,HD]          8 MiB
  __bf16* Kb  = (__bf16*)(ws + (24ull << 20));  // [B,H,S,HD]          8 MiB
  __bf16* Vtb = (__bf16*)(ws + (32ull << 20));  // [B,H,HD,S]          8 MiB
  __bf16* Ctx = (__bf16*)(ws + (40ull << 20));  // [4096][1024]        8 MiB
  float*  Tmp = (float* )(ws + (48ull << 20));  // [4096][1024] f32   16 MiB

  cvt_f32_bf16<<<(MM * DD / 4) / 256, 256, 0, stream>>>(x, Xb, MM * DD / 4);
  cvt_w_T<<<(DD * DD) / 256, 256, 0, stream>>>(wq, Wt + 0ull * DD * DD);
  cvt_w_T<<<(DD * DD) / 256, 256, 0, stream>>>(wk, Wt + 1ull * DD * DD);
  cvt_w_T<<<(DD * DD) / 256, 256, 0, stream>>>(wv, Wt + 2ull * DD * DD);
  cvt_w_T<<<(DD * DD) / 256, 256, 0, stream>>>(wo, Wt + 3ull * DD * DD);

  gemm_qkv<<<dim3(DD / 64, MM / 128, 3), 256, 0, stream>>>(Xb, Wt, Qb, Kb, Vtb);
  attn_kernel<<<dim3(SS / 64, BB * HH), 128, 0, stream>>>(Qb, Kb, Vtb, Ctx);
  gemm_out<<<dim3(DD / 64, MM / 128), 256, 0, stream>>>(Ctx, Wt + 3ull * DD * DD,
                                                        bo, x, Tmp);
  layernorm_kernel<<<MM, 256, 0, stream>>>(Tmp, gam, bet, outp);
}
